// PixelContrastCrossEntropyLoss_60876866453606
// MI455X (gfx1250) — matrix operand order repaired and checked by reading, not compiled, not run on hardware
//
#include <hip/hip_runtime.h>
#include <hip/hip_bf16.h>

typedef __attribute__((ext_vector_type(16))) _Float16 v16h;
typedef __attribute__((ext_vector_type(8)))  float    v8f;

#define A_DIM 16
#define V_DIM 512
#define C_DIM 256
#define N_DIM 8192          // V*A
#define NTILE 512           // N / 16
#define NPAIR 256           // NTILE / 2
#define KSTEPS 8            // 256 / 32
#define ROWTILES_PER_BLOCK 4
#define NBLOCKS (NTILE / ROWTILES_PER_BLOCK)   // 128

union U32B { uint4 q[2]; v16h h; };

// ---------------------------------------------------------------------------
// Kernel 1: gather embed rows, convert fp32 -> f16 into workspace, compute
// per-row squared norm of the *f16-rounded* values (== WMMA diag product),
// and zero the output scalar.
// Row n = v*16 + a  ->  feats[sample_batch[a]][sample_pixel[a][v]][:]
// ---------------------------------------------------------------------------
__global__ void gather_f16_kernel(const float* __restrict__ feats,
                                  const int* __restrict__ sbatch,
                                  const int* __restrict__ spixel,
                                  _Float16* __restrict__ featH,
                                  float* __restrict__ rowDiag,
                                  float* __restrict__ out) {
    int n = blockIdx.x;          // 0..8191
    int c = threadIdx.x;         // 0..255
    int lane = c & 31, wave = c >> 5;
    int v = n >> 4;
    int a = n & 15;
    int b = sbatch[a];
    int p = spixel[a * V_DIM + v];
    float val = feats[((size_t)b * 16384 + (size_t)p) * C_DIM + c];
    _Float16 h = (_Float16)val;
    featH[(size_t)n * C_DIM + c] = h;

    // block-reduce sum of squares of the f16-rounded values
    float hv = (float)h;
    float sq = hv * hv;
    #pragma unroll
    for (int off = 1; off < 32; off <<= 1) sq += __shfl_xor(sq, off, 32);
    __shared__ float part[8];
    if (lane == 0) part[wave] = sq;
    __syncthreads();
    if (c == 0) {
        float s = 0.0f;
        #pragma unroll
        for (int w = 0; w < 8; ++w) s += part[w];
        rowDiag[n] = s;
        if (n == 0) *out = 0.0f;
    }
}

// ---------------------------------------------------------------------------
// Kernel 2: fused contrastive loss.
// Block = 64 rows (4 row tiles). Wave pair (2t, 2t+1) owns row tile t and
// splits the 256 J-tile-pairs by parity, so waves of equal parity across
// pairs issue identical B loads (4-way WGP$ reuse). Two independent WMMA
// accumulator chains per wave for ILP.
// Sweep 1: online rowmax + neg-exp-sum + label-masked logit sum (incl diag).
// Sweep 2: sum over label-equal of log(exp(x-M)+negsum) (incl diag).
// Diagonal contribution removed analytically at the end via rowDiag.
// ---------------------------------------------------------------------------
__global__ __launch_bounds__(256)
void contrast_loss_kernel(const _Float16* __restrict__ featH,
                          const float* __restrict__ rowDiag,
                          const int* __restrict__ slabel,
                          float* __restrict__ out) {
    const int tid    = threadIdx.x;
    const int lane   = tid & 31;
    const int wave   = tid >> 5;       // 0..7
    const int pair   = wave >> 1;      // 0..3   (row tile within block)
    const int parity = wave & 1;       // J-range split within the pair
    const int Ig     = blockIdx.x * ROWTILES_PER_BLOCK + pair;   // global row tile
    const int n      = lane & 15;      // C col / A row / B col
    const int hi     = lane >> 4;      // half-wave select

    __shared__ int   labs[16];
    __shared__ float sM [8][16];
    __shared__ float sS [8][16];
    __shared__ float sP [8][16];
    __shared__ float s2a[8][16];

    if (tid < 16) labs[tid] = slabel[tid];
    __syncthreads();

    // base_mask is 16-periodic: identical for every 16x16 tile.
    // C layout: accumulator element r in this lane is (row = r + 8*hi, col = n).
    bool eqr[8];
    #pragma unroll
    for (int r = 0; r < 8; ++r) eqr[r] = (labs[r + 8 * hi] == labs[n]);

    // ---- Load A fragments (rows Ig*16 .. Ig*16+15, all K) once. ----
    // 16-bit A 16x32 layout: lane m = lane%16; VGPR v<4: K = k0 + 8*hi + 2v,
    // VGPR v>=4: K = k0 + 16 + 8*hi + 2(v-4)  -> two contiguous 16B chunks.
    v16h afrag[KSTEPS];
    {
        const _Float16* rowp = featH + (size_t)(Ig * 16 + n) * C_DIM;
        #pragma unroll
        for (int ks = 0; ks < KSTEPS; ++ks) {
            const int k0 = ks * 32;
            U32B u;
            u.q[0] = *(const uint4*)(rowp + k0 + 8 * hi);
            u.q[1] = *(const uint4*)(rowp + k0 + 16 + 8 * hi);
            afrag[ks] = u.h;
        }
    }

    // B fragment base for this lane: column n of tile J lives in feat row J*16+n.
    // 16-bit B 32x16 layout: j<8: K = k0 + 8*hi + j, j>=8: K = k0 + 16 + 8*hi + (j-8).
    const _Float16* bbase = featH + (size_t)n * C_DIM + 8 * hi;

    // ======================= Sweep 1 =======================
    float mr[8], sr[8], pd[8];
    #pragma unroll
    for (int r = 0; r < 8; ++r) { mr[r] = -__builtin_inff(); sr[r] = 0.0f; pd[r] = 0.0f; }

    for (int jp = parity; jp < NPAIR; jp += 2) {
        const _Float16* rowp0 = bbase + (size_t)jp * 32 * C_DIM;
        const _Float16* rowp1 = rowp0 + (size_t)16 * C_DIM;
        v8f c0 = {}, c1 = {};
        #pragma unroll
        for (int ks = 0; ks < KSTEPS; ++ks) {
            const int k0 = ks * 32;
            U32B u0, u1;
            u0.q[0] = *(const uint4*)(rowp0 + k0);
            u0.q[1] = *(const uint4*)(rowp0 + k0 + 16);
            u1.q[0] = *(const uint4*)(rowp1 + k0);
            u1.q[1] = *(const uint4*)(rowp1 + k0 + 16);
            c0 = __builtin_amdgcn_wmma_f32_16x16x32_f16(false, afrag[ks], false, u0.h,
                                                        (short)0, c0, false, false);
            c1 = __builtin_amdgcn_wmma_f32_16x16x32_f16(false, afrag[ks], false, u1.h,
                                                        (short)0, c1, false, false);
        }
        #pragma unroll
        for (int t = 0; t < 2; ++t) {
            const v8f& c = t ? c1 : c0;
            #pragma unroll
            for (int r = 0; r < 8; ++r) {
                float x  = c[r] * 10.0f;                // /TEMPERATURE
                float mn = fmaxf(mr[r], x);
                float sadd = eqr[r] ? 0.0f : __expf(x - mn);
                sr[r] = sr[r] * __expf(mr[r] - mn) + sadd;
                mr[r] = mn;
                pd[r] += eqr[r] ? x : 0.0f;             // diag removed later
            }
        }
    }

    // reduce across the 16 lanes of each half-wave (one row per (r,hi))
    #pragma unroll
    for (int off = 1; off < 16; off <<= 1) {
        #pragma unroll
        for (int r = 0; r < 8; ++r) {
            float mo = __shfl_xor(mr[r], off, 16);
            float so = __shfl_xor(sr[r], off, 16);
            float po = __shfl_xor(pd[r], off, 16);
            float mn = fmaxf(mr[r], mo);
            sr[r] = sr[r] * __expf(mr[r] - mn) + so * __expf(mo - mn);
            mr[r] = mn;
            pd[r] += po;
        }
    }
    if ((lane & 15) == 0) {
        #pragma unroll
        for (int r = 0; r < 8; ++r) {
            int row = r + 8 * hi;
            sM[wave][row] = mr[r];
            sS[wave][row] = sr[r];
            sP[wave][row] = pd[r];
        }
    }
    __syncthreads();

    // merge the wave pair's partials; every lane reads the broadcast values
    float Mr2[8], Sr2[8];
    {
        const int w0 = 2 * pair, w1 = w0 + 1;
        #pragma unroll
        for (int r = 0; r < 8; ++r) {
            int row = r + 8 * hi;
            float m0 = sM[w0][row], s0 = sS[w0][row];
            float m1 = sM[w1][row], s1 = sS[w1][row];
            float mn = fmaxf(m0, m1);
            Mr2[r] = mn;
            Sr2[r] = s0 * __expf(m0 - mn) + s1 * __expf(m1 - mn);
        }
    }

    // ======================= Sweep 2 =======================
    float s2[8];
    #pragma unroll
    for (int r = 0; r < 8; ++r) s2[r] = 0.0f;

    for (int jp = parity; jp < NPAIR; jp += 2) {
        const _Float16* rowp0 = bbase + (size_t)jp * 32 * C_DIM;
        const _Float16* rowp1 = rowp0 + (size_t)16 * C_DIM;
        v8f c0 = {}, c1 = {};
        #pragma unroll
        for (int ks = 0; ks < KSTEPS; ++ks) {
            const int k0 = ks * 32;
            U32B u0, u1;
            u0.q[0] = *(const uint4*)(rowp0 + k0);
            u0.q[1] = *(const uint4*)(rowp0 + k0 + 16);
            u1.q[0] = *(const uint4*)(rowp1 + k0);
            u1.q[1] = *(const uint4*)(rowp1 + k0 + 16);
            c0 = __builtin_amdgcn_wmma_f32_16x16x32_f16(false, afrag[ks], false, u0.h,
                                                        (short)0, c0, false, false);
            c1 = __builtin_amdgcn_wmma_f32_16x16x32_f16(false, afrag[ks], false, u1.h,
                                                        (short)0, c1, false, false);
        }
        #pragma unroll
        for (int t = 0; t < 2; ++t) {
            const v8f& c = t ? c1 : c0;
            #pragma unroll
            for (int r = 0; r < 8; ++r) {
                if (eqr[r]) {
                    float x = c[r] * 10.0f;
                    s2[r] += __logf(__expf(x - Mr2[r]) + Sr2[r]);   // diag removed later
                }
            }
        }
    }
    #pragma unroll
    for (int off = 1; off < 16; off <<= 1) {
        #pragma unroll
        for (int r = 0; r < 8; ++r) s2[r] += __shfl_xor(s2[r], off, 16);
    }
    if ((lane & 15) == 0) {
        #pragma unroll
        for (int r = 0; r < 8; ++r) s2a[wave][r + 8 * hi] = s2[r];
    }
    __syncthreads();

    // ---- Final: one thread per (row tile, row) = 64 threads ----
    if (tid < 16 * ROWTILES_PER_BLOCK) {
        const int pt  = tid >> 4;          // row tile within block
        const int row = tid & 15;
        const int w0 = 2 * pt, w1 = w0 + 1;
        float m0 = sM[w0][row], s0 = sS[w0][row];
        float m1 = sM[w1][row], s1 = sS[w1][row];
        float M  = fmaxf(m0, m1);
        float S  = s0 * __expf(m0 - M) + s1 * __expf(m1 - M);
        float Pd = sP[w0][row] + sP[w1][row];
        float S2 = s2a[w0][row] + s2a[w1][row];

        // remove the diagonal (self) contribution analytically
        const int gi = (blockIdx.x * ROWTILES_PER_BLOCK + pt) * 16 + row;
        float xd = 10.0f * rowDiag[gi];
        Pd -= xd;
        S2 -= __logf(__expf(xd - M) + S);

        int cnt = 0;
        for (int a = 0; a < 16; ++a) cnt += (labs[a] == labs[row]);
        float P = (float)(V_DIM * cnt - 1);          // positive-mask row sum

        // mean_log_prob_pos = (sum_pos x - M*P - sum_pos log(exp(x-M)+negsum)) / P
        float mlpp = (Pd - M * P - S2) / P;
        // out = LOSS_WEIGHT * mean_i( -(T/baseT) * mlpp_i )
        const float COEF = -(0.1f / 0.07f) * 0.1f / (float)N_DIM;
        atomicAdd(out, COEF * mlpp);
    }
}

// ---------------------------------------------------------------------------
extern "C" void kernel_launch(void* const* d_in, const int* in_sizes, int n_in,
                              void* d_out, int out_size, void* d_ws, size_t ws_size,
                              hipStream_t stream) {
    const float* feats  = (const float*)d_in[0];
    const int*   sbatch = (const int*)d_in[1];
    const int*   spixel = (const int*)d_in[2];
    const int*   slabel = (const int*)d_in[3];
    float*       out    = (float*)d_out;
    _Float16*    featH   = (_Float16*)d_ws;                       // 4 MB
    float*       rowDiag = (float*)((char*)d_ws + (size_t)N_DIM * C_DIM * 2); // 32 KB

    gather_f16_kernel<<<N_DIM, C_DIM, 0, stream>>>(feats, sbatch, spixel,
                                                   featH, rowDiag, out);
    contrast_loss_kernel<<<NBLOCKS, 256, 0, stream>>>(featH, rowDiag, slabel, out);
    (void)in_sizes; (void)n_in; (void)out_size; (void)ws_size;
}
